// RaggedAttention_49486613185349
// MI455X (gfx1250) — matrix operand order repaired and checked by reading, not compiled, not run on hardware
//
#include <hip/hip_runtime.h>

// ---------------------------------------------------------------------------
// Ragged relative-position attention for MI455X (gfx1250, wave32, WMMA).
// All matmuls use v_wmma_f32_16x16x32_f16 (codegen-verified builtin).
// Streaming operands (relkeys/relvals, 604 MB single-pass) use non-temporal
// loads so they don't evict the reused q/k/v/P working set from the 192 MB L2.
// ---------------------------------------------------------------------------

#define D_MODELX 512
#define N_HEADX  8
#define HEAD_DIMX 64
#define BBX 8
#define TTX 384
#define N_TOKENSX 2048
#define NPACKX (BBX * TTX)   // 3072

typedef __attribute__((ext_vector_type(16))) _Float16 v16h;
typedef __attribute__((ext_vector_type(8)))  float    v8f;

#define DEVINL static __device__ __forceinline__

// D = A*B + C, 16x16 f32 accum, A/B 16x32 / 32x16 f16
DEVINL v8f wmma16(v16h a, v16h b, v8f c) {
  return __builtin_amdgcn_wmma_f32_16x16x32_f16(
      /*neg_a=*/false, a, /*neg_b=*/false, b,
      /*c_mod=*/(short)0, c, /*reuse_a=*/false, /*reuse_b=*/false);
}

// A fragment 16x32 (MxK), row-major source: element (m,k) = src[m*ld + k].
// Contiguous-k layout used here: lane = {kh,m[3:0]}, VGPR j -> K = kh*16 + 2j, 2j+1.
// Rows >= `rows` are zero (used to pad M=8 head dimension to 16).
DEVINL v16h load_a_h(const _Float16* __restrict__ src, size_t ld, int lane, int rows) {
  int m = lane & 15, kh = lane >> 4;
  v16h f;
#pragma unroll
  for (int j = 0; j < 8; ++j) {
    int kb = kh * 16 + 2 * j;
    if (m < rows) {
      f[2 * j]     = src[(size_t)m * ld + kb];
      f[2 * j + 1] = src[(size_t)m * ld + kb + 1];
    } else {
      f[2 * j] = (_Float16)0.0f;
      f[2 * j + 1] = (_Float16)0.0f;
    }
  }
  return f;
}

// Same, converting from f32 source on load.
DEVINL v16h load_a_f(const float* __restrict__ src, size_t ld, int lane) {
  int m = lane & 15, kh = lane >> 4;
  v16h f;
#pragma unroll
  for (int j = 0; j < 8; ++j) {
    int kb = kh * 16 + 2 * j;
    f[2 * j]     = (_Float16)src[(size_t)m * ld + kb];
    f[2 * j + 1] = (_Float16)src[(size_t)m * ld + kb + 1];
  }
  return f;
}

// B fragment 32x16 (KxN), N-major source: element (k,n) = src[n*ld + k].
// Contiguous-k layout used here: lane = {kh,n[3:0]}, VGPR j -> K = kh*16 + 2j, 2j+1.
DEVINL v16h load_b_nmaj_h(const _Float16* __restrict__ src, size_t ld, int lane) {
  int n = lane & 15, kh = lane >> 4;
  v16h f;
#pragma unroll
  for (int j = 0; j < 8; ++j) {
    int k = kh * 16 + 2 * j;
    f[2 * j]     = src[(size_t)n * ld + k];
    f[2 * j + 1] = src[(size_t)n * ld + k + 1];
  }
  return f;
}

// Non-temporal f32 variant: streamed-once operand, TH=NT so it bypasses
// long-term L2 residency (rel tensors are never re-read).
DEVINL v16h load_b_nmaj_f_nt(const float* __restrict__ src, size_t ld, int lane) {
  int n = lane & 15, kh = lane >> 4;
  v16h f;
#pragma unroll
  for (int j = 0; j < 8; ++j) {
    int k = kh * 16 + 2 * j;
    f[2 * j]     = (_Float16)__builtin_nontemporal_load(&src[(size_t)n * ld + k]);
    f[2 * j + 1] = (_Float16)__builtin_nontemporal_load(&src[(size_t)n * ld + k + 1]);
  }
  return f;
}

// B fragment, K-major source: element (k,n) = src[k*ld + n].
DEVINL v16h load_b_kmaj_h(const _Float16* __restrict__ src, size_t ld, int lane) {
  int n = lane & 15, kh = lane >> 4;
  v16h f;
#pragma unroll
  for (int j = 0; j < 8; ++j) {
    int k = kh * 16 + 2 * j;
    f[2 * j]     = src[(size_t)k * ld + n];
    f[2 * j + 1] = src[(size_t)(k + 1) * ld + n];
  }
  return f;
}

DEVINL v16h load_b_kmaj_f_nt(const float* __restrict__ src, size_t ld, int lane) {
  int n = lane & 15, kh = lane >> 4;
  v16h f;
#pragma unroll
  for (int j = 0; j < 8; ++j) {
    int k = kh * 16 + 2 * j;
    f[2 * j]     = (_Float16)__builtin_nontemporal_load(&src[(size_t)k * ld + n]);
    f[2 * j + 1] = (_Float16)__builtin_nontemporal_load(&src[(size_t)(k + 1) * ld + n]);
  }
  return f;
}

// ---------------------------------------------------------------------------
// K1: fp32 -> f16 conversion (weights)
__global__ void k_f32_to_f16(const float* __restrict__ src, _Float16* __restrict__ dst, int n) {
  int i = blockIdx.x * blockDim.x + threadIdx.x;
  if (i < n) dst[i] = (_Float16)src[i];
}

// K2: gather ragged tokens into packed layout, convert to f16
__global__ void k_gather_x(const float* __restrict__ x, const int* __restrict__ idx,
                           _Float16* __restrict__ xp) {
  int i = blockIdx.x * blockDim.x + threadIdx.x;   // over NPACK*D_MODEL
  int r = i >> 9, c = i & (D_MODELX - 1);
  xp[i] = (_Float16)x[(size_t)idx[r] * D_MODELX + c];
}

// K3: projection GEMM  out[b,h,s,d] = xp[r,:] . W[c,:] + bias[c]   (c = h*64+d, r = b*T+s)
// One wave per 16x16 output tile; K=512 in 16 WMMA steps of k=32.
__global__ void k_qkv_gemm(const _Float16* __restrict__ xp, const _Float16* __restrict__ w,
                           const float* __restrict__ bias, _Float16* __restrict__ out) {
  int r0 = blockIdx.x * 16, c0 = blockIdx.y * 16;
  int lane = threadIdx.x;
  v8f acc = {};
#pragma unroll
  for (int k0 = 0; k0 < D_MODELX; k0 += 32) {
    v16h a   = load_a_h(xp + (size_t)r0 * D_MODELX + k0, D_MODELX, lane, 16);
    v16h bfr = load_b_nmaj_h(w + (size_t)c0 * D_MODELX + k0, D_MODELX, lane);
    acc = wmma16(a, bfr, acc);
  }
  int n = lane & 15, mh = lane >> 4;
  int c = c0 + n, h = c >> 6, d = c & 63;
  float bv = bias[c];
#pragma unroll
  for (int j = 0; j < 8; ++j) {
    int r = r0 + mh * 8 + j;
    int bb = r / TTX, s = r % TTX;
    out[(((size_t)bb * N_HEADX + h) * TTX + s) * HEAD_DIMX + d] = (_Float16)(acc[j] + bv);
  }
}

// K4: rel logits per (b,s):  att[b,h,s,t] = scale * q[b,h,s,:] . relkeys[b,s,t,:]
// M = heads (8 valid of 16), N = t tile, K = 64. relkeys streamed NT.
__global__ void k_rel_logits(const _Float16* __restrict__ q16, const float* __restrict__ relkeys,
                             _Float16* __restrict__ att) {
  int bs = blockIdx.x;            // b*T + s
  int t0 = blockIdx.y * 16;
  int b = bs / TTX, s = bs % TTX;
  int lane = threadIdx.x;
  const float scale = 0.125f;     // 1/sqrt(64)
  const float* relbase = relkeys + ((size_t)bs * TTX + t0) * HEAD_DIMX;
  __builtin_prefetch(relbase + 16 * HEAD_DIMX, 0, 1);   // global_prefetch next tile
  const _Float16* qbase = q16 + ((size_t)b * N_HEADX * TTX + s) * HEAD_DIMX;
  v8f acc = {};
#pragma unroll
  for (int k0 = 0; k0 < HEAD_DIMX; k0 += 32) {
    v16h a   = load_a_h(qbase + k0, (size_t)TTX * HEAD_DIMX, lane, N_HEADX);
    v16h bfr = load_b_nmaj_f_nt(relbase + k0, HEAD_DIMX, lane);
    acc = wmma16(a, bfr, acc);
  }
  int n = lane & 15, mh = lane >> 4;
#pragma unroll
  for (int j = 0; j < 8; ++j) {
    int h = mh * 8 + j;
    if (h < N_HEADX)
      att[(((size_t)b * N_HEADX + h) * TTX + s) * TTX + t0 + n] = (_Float16)(acc[j] * scale);
  }
}

// K5: per (b,h,s-tile): logits = scale*Q@K^T + rel; mask; softmax; P back to att (f16);
//     fused Y1 = P @ V_h -> y_ws.   256 threads = 8 waves.
__global__ void k_attn_softmax(const _Float16* __restrict__ q16, const _Float16* __restrict__ k16,
                               const _Float16* __restrict__ vbuf, const int* __restrict__ pb,
                               _Float16* __restrict__ att, float* __restrict__ y_ws) {
  __shared__ float sm[16][TTX];     // 24 KB row block of logits/probs
  __shared__ float red[16][16];
  int s0 = blockIdx.x * 16;
  int h  = blockIdx.y;
  int b  = blockIdx.z;
  int tid = threadIdx.x;
  int wave = tid >> 5, lane = tid & 31;
  const float scale = 0.125f;
  size_t headoff = ((size_t)b * N_HEADX + h) * TTX;   // row base for [.,64] q/k/v arrays
  const int* pbb = pb + b * TTX;

  // Q A-fragments are invariant across t-tiles: load once per wave.
  v16h qa0 = load_a_h(q16 + (headoff + s0) * HEAD_DIMX +  0, HEAD_DIMX, lane, 16);
  v16h qa1 = load_a_h(q16 + (headoff + s0) * HEAD_DIMX + 32, HEAD_DIMX, lane, 16);

  // ---- phase 1: logits via WMMA, 3 t-tiles per wave -----------------------
  for (int tt = wave; tt < TTX / 16; tt += 8) {
    int t0 = tt * 16;
    v8f acc = {};
    v16h b0 = load_b_nmaj_h(k16 + (headoff + t0) * HEAD_DIMX +  0, HEAD_DIMX, lane);
    v16h b1 = load_b_nmaj_h(k16 + (headoff + t0) * HEAD_DIMX + 32, HEAD_DIMX, lane);
    acc = wmma16(qa0, b0, acc);
    acc = wmma16(qa1, b1, acc);
    int n = lane & 15, mh = lane >> 4;
#pragma unroll
    for (int j = 0; j < 8; ++j) {
      int m = mh * 8 + j;
      int s = s0 + m, t = t0 + n;
      float rel = (float)att[(headoff + s) * TTX + t];
      float vlog = acc[j] * scale + rel;
      if (pbb[s] != pbb[t]) vlog = -1e9f;   // block-diagonal ragged mask
      sm[m][t] = vlog;
    }
  }
  __syncthreads();

  // ---- phase 2: row softmax (16 rows x 16 threads each) -------------------
  {
    int r = tid >> 4, l = tid & 15;
    float mx = -3.4e38f;
    for (int t = l; t < TTX; t += 16) mx = fmaxf(mx, sm[r][t]);
    red[r][l] = mx;
    __syncthreads();
    if (l == 0) {
      float m2 = red[r][0];
#pragma unroll
      for (int i = 1; i < 16; ++i) m2 = fmaxf(m2, red[r][i]);
      red[r][0] = m2;
    }
    __syncthreads();
    float rowmax = red[r][0];
    __syncthreads();
    float psum = 0.0f;
    for (int t = l; t < TTX; t += 16) {
      float e = __expf(sm[r][t] - rowmax);
      sm[r][t] = e;
      psum += e;
    }
    red[r][l] = psum;
    __syncthreads();
    if (l == 0) {
      float s2 = 0.0f;
#pragma unroll
      for (int i = 0; i < 16; ++i) s2 += red[r][i];
      red[r][0] = s2;
    }
    __syncthreads();
    float inv = 1.0f / red[r][0];
    for (int t = l; t < TTX; t += 16) {
      float p = sm[r][t] * inv;
      sm[r][t] = p;
      att[(headoff + s0 + r) * TTX + t] = (_Float16)p;   // in-place P for K6
    }
  }
  __syncthreads();

  // ---- phase 3: Y1 = P @ V_h, waves 0..3 handle the four 16-wide d tiles --
  if (wave < 4) {
    int d0 = wave * 16;
    v8f acc = {};
    for (int t0 = 0; t0 < TTX; t0 += 32) {
      v16h a   = load_a_f(&sm[0][0] + t0, TTX, lane);
      v16h bfr = load_b_kmaj_h(vbuf + (headoff + t0) * HEAD_DIMX + d0, HEAD_DIMX, lane);
      acc = wmma16(a, bfr, acc);
    }
    int n = lane & 15, mh = lane >> 4;
#pragma unroll
    for (int j = 0; j < 8; ++j) {
      int s = s0 + mh * 8 + j;
      y_ws[((size_t)b * TTX + s) * D_MODELX + h * HEAD_DIMX + d0 + n] = acc[j];
    }
  }
}

// K6: per (b,s): Y2[h,d] = sum_t P[b,h,s,t] * relvals[b,s,t,d]; y_ws += Y2.
// relvals streamed NT (read exactly once).
__global__ void k_rel_out(const _Float16* __restrict__ att, const float* __restrict__ relvals,
                          float* __restrict__ y_ws) {
  int bs = blockIdx.x;
  int d0 = blockIdx.y * 16;
  int b = bs / TTX, s = bs % TTX;
  int lane = threadIdx.x;
  const float* rvbase = relvals + (size_t)bs * TTX * HEAD_DIMX + d0;
  const _Float16* pbase = att + ((size_t)b * N_HEADX * TTX + s) * TTX;  // row stride T*T per head
  v8f acc = {};
  for (int t0 = 0; t0 < TTX; t0 += 32) {
    __builtin_prefetch(rvbase + (size_t)(t0 + 32) * HEAD_DIMX, 0, 1);
    v16h a   = load_a_h(pbase + t0, (size_t)TTX * TTX, lane, N_HEADX);
    v16h bfr = load_b_kmaj_f_nt(rvbase + (size_t)t0 * HEAD_DIMX, HEAD_DIMX, lane);
    acc = wmma16(a, bfr, acc);
  }
  int n = lane & 15, mh = lane >> 4;
#pragma unroll
  for (int j = 0; j < 8; ++j) {
    int h = mh * 8 + j;
    if (h < N_HEADX) {
      size_t o = (size_t)bs * D_MODELX + h * HEAD_DIMX + d0 + n;
      y_ws[o] += acc[j];     // exclusive writer per (b,s,h,d) -> no atomics needed
    }
  }
}

// K7: output projection proj = yh @ Wp^T + bp (A converted f32->f16 on load)
__global__ void k_proj(const float* __restrict__ y_ws, const _Float16* __restrict__ wp,
                       const float* __restrict__ bp, float* __restrict__ proj) {
  int r0 = blockIdx.x * 16, c0 = blockIdx.y * 16;
  int lane = threadIdx.x;
  v8f acc = {};
#pragma unroll
  for (int k0 = 0; k0 < D_MODELX; k0 += 32) {
    v16h a   = load_a_f(y_ws + (size_t)r0 * D_MODELX + k0, D_MODELX, lane);
    v16h bfr = load_b_nmaj_h(wp + (size_t)c0 * D_MODELX + k0, D_MODELX, lane);
    acc = wmma16(a, bfr, acc);
  }
  int n = lane & 15, mh = lane >> 4;
  float bb = bp[c0 + n];
#pragma unroll
  for (int j = 0; j < 8; ++j)
    proj[(size_t)(r0 + mh * 8 + j) * D_MODELX + c0 + n] = acc[j] + bb;
}

// K8: scatter back to ragged token order
__global__ void k_scatter(const float* __restrict__ proj, const int* __restrict__ inv,
                          float* __restrict__ out) {
  int i = blockIdx.x * blockDim.x + threadIdx.x;
  if (i >= N_TOKENSX * D_MODELX) return;
  int r = i >> 9, c = i & (D_MODELX - 1);
  out[i] = proj[(size_t)inv[r] * D_MODELX + c];
}

// ---------------------------------------------------------------------------
extern "C" void kernel_launch(void* const* d_in, const int* in_sizes, int n_in,
                              void* d_out, int out_size, void* d_ws, size_t ws_size,
                              hipStream_t stream) {
  (void)in_sizes; (void)n_in; (void)out_size; (void)ws_size;
  const float* x        = (const float*)d_in[0];
  const float* relkeys  = (const float*)d_in[1];
  const float* relvals  = (const float*)d_in[2];
  const float* Wq       = (const float*)d_in[3];
  const float* bq       = (const float*)d_in[4];
  const float* Wk       = (const float*)d_in[5];
  const float* bk       = (const float*)d_in[6];
  const float* Wv       = (const float*)d_in[7];
  const float* bv       = (const float*)d_in[8];
  const float* Wp       = (const float*)d_in[9];
  const float* bp       = (const float*)d_in[10];
  const int* pp_index   = (const int*)d_in[11];
  const int* pp_batch   = (const int*)d_in[12];
  const int* pp_inv     = (const int*)d_in[13];
  float* out = (float*)d_out;

  // workspace carve-up (~45 MB total)
  char* ws = (char*)d_ws;
  size_t off = 0;
  auto take = [&](size_t bytes) -> void* {
    void* p = ws + off;
    off += (bytes + 255) & ~(size_t)255;
    return p;
  };
  const size_t WSZ = (size_t)D_MODELX * D_MODELX;
  _Float16* wq16 = (_Float16*)take(WSZ * 2);
  _Float16* wk16 = (_Float16*)take(WSZ * 2);
  _Float16* wv16 = (_Float16*)take(WSZ * 2);
  _Float16* wp16 = (_Float16*)take(WSZ * 2);
  _Float16* xp16 = (_Float16*)take((size_t)NPACKX * D_MODELX * 2);
  _Float16* q16  = (_Float16*)take((size_t)NPACKX * D_MODELX * 2);
  _Float16* k16  = (_Float16*)take((size_t)NPACKX * D_MODELX * 2);
  _Float16* v16b = (_Float16*)take((size_t)NPACKX * D_MODELX * 2);
  _Float16* att  = (_Float16*)take((size_t)BBX * N_HEADX * TTX * TTX * 2);  // 18.9 MB
  float* y_ws    = (float*)take((size_t)NPACKX * D_MODELX * 4);
  float* proj    = (float*)take((size_t)NPACKX * D_MODELX * 4);

  // 1: weights -> f16
  int wblocks = (int)((WSZ + 255) / 256);
  k_f32_to_f16<<<wblocks, 256, 0, stream>>>(Wq, wq16, (int)WSZ);
  k_f32_to_f16<<<wblocks, 256, 0, stream>>>(Wk, wk16, (int)WSZ);
  k_f32_to_f16<<<wblocks, 256, 0, stream>>>(Wv, wv16, (int)WSZ);
  k_f32_to_f16<<<wblocks, 256, 0, stream>>>(Wp, wp16, (int)WSZ);

  // 2: gather + convert activations
  k_gather_x<<<(NPACKX * D_MODELX) / 256, 256, 0, stream>>>(x, pp_index, xp16);

  // 3: Q/K/V projections (WMMA)
  dim3 gGemm(NPACKX / 16, D_MODELX / 16);
  k_qkv_gemm<<<gGemm, 32, 0, stream>>>(xp16, wq16, bq, q16);
  k_qkv_gemm<<<gGemm, 32, 0, stream>>>(xp16, wk16, bk, k16);
  k_qkv_gemm<<<gGemm, 32, 0, stream>>>(xp16, wv16, bv, v16b);

  // 4: rel-key logits (WMMA, streams relkeys 302 MB once, NT)
  dim3 gRel(NPACKX, TTX / 16);
  k_rel_logits<<<gRel, 32, 0, stream>>>(q16, relkeys, att);

  // 5: main logits + mask + softmax + Y1 = P@V (WMMA)
  dim3 gAttn(TTX / 16, N_HEADX, BBX);
  k_attn_softmax<<<gAttn, 256, 0, stream>>>(q16, k16, v16b, pp_batch, att, y_ws);

  // 6: Y2 = P @ relvals (WMMA, streams relvals 302 MB once, NT)
  dim3 gRo(NPACKX, HEAD_DIMX / 16);
  k_rel_out<<<gRo, 32, 0, stream>>>(att, relvals, y_ws);

  // 7: output projection (WMMA)
  k_proj<<<gGemm, 32, 0, stream>>>(y_ws, wp16, bp, proj);

  // 8: scatter to ragged order
  k_scatter<<<(N_TOKENSX * D_MODELX + 255) / 256, 256, 0, stream>>>(proj, pp_inv, out);
}
